// MessagePassingLayer_86543591015027
// MI455X (gfx1250) — matrix operand order repaired and checked by reading, not compiled, not run on hardware
//
#include <hip/hip_runtime.h>

#define N_NODES 50000
#define N_EDGES 800000

typedef _Float16 v16h __attribute__((ext_vector_type(16)));
typedef _Float16 v8h  __attribute__((ext_vector_type(8)));
typedef float    v8f  __attribute__((ext_vector_type(8)));
typedef float    v4f  __attribute__((ext_vector_type(4)));

__device__ __forceinline__ v8h cvt8(const float* __restrict__ p, float s) {
    v4f a = *(const v4f*)p;
    v4f b = *(const v4f*)(p + 4);
    v8h r;
    r[0] = (_Float16)(a.x * s); r[1] = (_Float16)(a.y * s);
    r[2] = (_Float16)(a.z * s); r[3] = (_Float16)(a.w * s);
    r[4] = (_Float16)(b.x * s); r[5] = (_Float16)(b.y * s);
    r[6] = (_Float16)(b.z * s); r[7] = (_Float16)(b.w * s);
    return r;
}

__device__ __forceinline__ v16h cat16(v8h a, v8h b) {
    return __builtin_shufflevector(a, b, 0,1,2,3,4,5,6,7,8,9,10,11,12,13,14,15);
}

__device__ __forceinline__ float gelu_tanh(float x) {
    float u = 0.7978845608028654f * (x + 0.044715f * x * x * x);
    return 0.5f * x * (1.0f + tanhf(u));
}

__device__ __forceinline__ void lds_fence() {
    __builtin_amdgcn_wave_barrier();
    asm volatile("s_wait_dscnt 0" ::: "memory");
    __builtin_amdgcn_wave_barrier();
}

// Computes Y = LN-input (pre-norm) for a 16-row tile:
//   X[16x192] = [p0*s0 | p1*s1 | p2*s2] per row (row r uses caller-provided per-lane ptrs)
//   H = gelu(X @ W1 + b1)  [16x128],  Y = H @ W2 + b2  [16x64]  -> sY row-major
// lane&15 selects the tile row for A-fragments; both half-waves cooperate.
__device__ __forceinline__ void mlp_tile(
    const float* __restrict__ p0, float s0,
    const float* __restrict__ p1, float s1,
    const float* __restrict__ p2, float s2,
    const _Float16* __restrict__ W1T, const float* __restrict__ b1,
    const _Float16* __restrict__ W2T, const float* __restrict__ b2,
    _Float16* __restrict__ sH, float* __restrict__ sY, int lane)
{
    const int ln = lane & 15;
    const int g  = lane >> 4;

    v8f acc1[8];
#pragma unroll
    for (int t = 0; t < 8; ++t)
#pragma unroll
        for (int r = 0; r < 8; ++r) acc1[t][r] = 0.0f;

    // GEMM1: K=192 in 6 chunks of 32, N=128 in 8 tiles of 16
#pragma unroll
    for (int c = 0; c < 6; ++c) {
        const float* srcp = (c < 2) ? p0 : (c < 4) ? p1 : p2;
        const float  s    = (c < 2) ? s0 : (c < 4) ? s1 : s2;
        const int off = (c & 1) * 32 + g * 16;
        v8h alo = cvt8(srcp + off,      s);
        v8h ahi = cvt8(srcp + off + 8, s);
        v16h afrag = cat16(alo, ahi);
#pragma unroll
        for (int t = 0; t < 8; ++t) {
            const _Float16* wrow = W1T + (t * 16 + ln) * 192;
            v16h bfrag = *(const v16h*)(wrow + c * 32 + g * 16);
            acc1[t] = __builtin_amdgcn_wmma_f32_16x16x32_f16(
                false, afrag, false, bfrag, (short)0, acc1[t], false, false);
        }
    }

    // bias + GELU -> sH row-major [16][128] f16
#pragma unroll
    for (int t = 0; t < 8; ++t) {
        float bb = b1[t * 16 + ln];
#pragma unroll
        for (int r = 0; r < 8; ++r) {
            float h = gelu_tanh(acc1[t][r] + bb);
            sH[(r + 8 * g) * 128 + (t * 16 + ln)] = (_Float16)h;
        }
    }
    lds_fence();

    // GEMM2: K=128 in 4 chunks, N=64 in 4 tiles
    v8f acc2[4];
#pragma unroll
    for (int t = 0; t < 4; ++t)
#pragma unroll
        for (int r = 0; r < 8; ++r) acc2[t][r] = 0.0f;
#pragma unroll
    for (int c = 0; c < 4; ++c) {
        const _Float16* hrow = sH + ln * 128 + c * 32 + g * 16;
        v8h alo = *(const v8h*)(hrow);
        v8h ahi = *(const v8h*)(hrow + 8);
        v16h afrag = cat16(alo, ahi);
#pragma unroll
        for (int t = 0; t < 4; ++t) {
            const _Float16* wrow = W2T + (t * 16 + ln) * 128;
            v16h bfrag = *(const v16h*)(wrow + c * 32 + g * 16);
            acc2[t] = __builtin_amdgcn_wmma_f32_16x16x32_f16(
                false, afrag, false, bfrag, (short)0, acc2[t], false, false);
        }
    }

    // + b2 -> sY row-major [16][64] f32
#pragma unroll
    for (int t = 0; t < 4; ++t) {
        float bb = b2[t * 16 + ln];
#pragma unroll
        for (int r = 0; r < 8; ++r)
            sY[(r + 8 * g) * 64 + (t * 16 + ln)] = acc2[t][r] + bb;
    }
    lds_fence();
}

__device__ __forceinline__ void row_stats(const float* __restrict__ yrow,
                                          float& mu, float& rstd) {
    float sum = 0.f, sq = 0.f;
#pragma unroll
    for (int i = 0; i < 16; ++i) {
        v4f y = *(const v4f*)(yrow + i * 4);
        sum += y.x + y.y + y.z + y.w;
        sq  += y.x * y.x + y.y * y.y + y.z * y.z + y.w * y.w;
    }
    mu = sum * (1.0f / 64.0f);
    float var = sq * (1.0f / 64.0f) - mu * mu;
    rstd = rsqrtf(var + 1e-5f);
}

__global__ void __launch_bounds__(128)
edge_kernel(const float* __restrict__ ndata, const float* __restrict__ edata,
            const int* __restrict__ src, const int* __restrict__ dst,
            const float* __restrict__ eb1, const float* __restrict__ eb2,
            const float* __restrict__ egm, const float* __restrict__ ebeta,
            const _Float16* __restrict__ W1T, const _Float16* __restrict__ W2T,
            float* __restrict__ out_e,
            float* __restrict__ eu, float* __restrict__ ev,
            float* __restrict__ cu, float* __restrict__ cv)
{
    __shared__ alignas(32) _Float16 sH[4][16 * 128];
    __shared__ alignas(32) float    sY[4][16 * 64];
    const int w    = threadIdx.x >> 5;
    const int lane = threadIdx.x & 31;
    const int tile = blockIdx.x * 4 + w;
    if (tile >= N_EDGES / 16) return;
    const int ln = lane & 15;
    const int g  = lane >> 4;
    const int e  = tile * 16 + ln;
    const int si = src[e];
    const int di = dst[e];

    mlp_tile(ndata + si * 64, 1.0f,
             ndata + di * 64, 1.0f,
             edata + (long)e * 64, 1.0f,
             W1T, eb1, W2T, eb2, sH[w], sY[w], lane);

    // LayerNorm row ln; both half-lanes compute stats, split the 64-col write.
    const float* yrow = sY[w] + ln * 64;
    float mu, rstd;
    row_stats(yrow, mu, rstd);
    const int col0 = g * 32;
    float* orow = out_e + (long)e * 64;
#pragma unroll
    for (int j = 0; j < 8; ++j) {
        int col = col0 + j * 4;
        v4f y  = *(const v4f*)(yrow + col);
        v4f gg = *(const v4f*)(egm + col);
        v4f bb = *(const v4f*)(ebeta + col);
        v4f o;
        o.x = (y.x - mu) * rstd * gg.x + bb.x;
        o.y = (y.y - mu) * rstd * gg.y + bb.y;
        o.z = (y.z - mu) * rstd * gg.z + bb.z;
        o.w = (y.w - mu) * rstd * gg.w + bb.w;
        *(v4f*)(orow + col) = o;
        // fused scatter-sum for segment means (L2-resident accumulators)
        float* pu = eu + (long)si * 64 + col;
        float* pv = ev + (long)di * 64 + col;
        atomicAdd(pu + 0, o.x); atomicAdd(pu + 1, o.y);
        atomicAdd(pu + 2, o.z); atomicAdd(pu + 3, o.w);
        atomicAdd(pv + 0, o.x); atomicAdd(pv + 1, o.y);
        atomicAdd(pv + 2, o.z); atomicAdd(pv + 3, o.w);
    }
    if (g == 0) {
        atomicAdd(cu + si, 1.0f);
        atomicAdd(cv + di, 1.0f);
    }
}

__global__ void __launch_bounds__(128)
node_kernel(const float* __restrict__ ndata,
            const float* __restrict__ eu, const float* __restrict__ ev,
            const float* __restrict__ cu, const float* __restrict__ cv,
            const float* __restrict__ nb1, const float* __restrict__ nb2,
            const float* __restrict__ ngm, const float* __restrict__ nbeta,
            const _Float16* __restrict__ W1T, const _Float16* __restrict__ W2T,
            float* __restrict__ out_n)
{
    __shared__ alignas(32) _Float16 sH[4][16 * 128];
    __shared__ alignas(32) float    sY[4][16 * 64];
    const int w    = threadIdx.x >> 5;
    const int lane = threadIdx.x & 31;
    const int tile = blockIdx.x * 4 + w;
    if (tile >= N_NODES / 16) return;
    const int ln = lane & 15;
    const int g  = lane >> 4;
    const int nd = tile * 16 + ln;

    const float s0 = 1.0f / fmaxf(cu[nd], 1.0f);  // mean over src-edges
    const float s1 = 1.0f / fmaxf(cv[nd], 1.0f);  // mean over dst-edges

    mlp_tile(eu + (long)nd * 64, s0,
             ev + (long)nd * 64, s1,
             ndata + (long)nd * 64, 1.0f,
             W1T, nb1, W2T, nb2, sH[w], sY[w], lane);

    const float* yrow = sY[w] + ln * 64;
    float mu, rstd;
    row_stats(yrow, mu, rstd);
    const int col0 = g * 32;
    float* orow = out_n + (long)nd * 64;
#pragma unroll
    for (int j = 0; j < 8; ++j) {
        int col = col0 + j * 4;
        v4f y  = *(const v4f*)(yrow + col);
        v4f gg = *(const v4f*)(ngm + col);
        v4f bb = *(const v4f*)(nbeta + col);
        v4f o;
        o.x = (y.x - mu) * rstd * gg.x + bb.x;
        o.y = (y.y - mu) * rstd * gg.y + bb.y;
        o.z = (y.z - mu) * rstd * gg.z + bb.z;
        o.w = (y.w - mu) * rstd * gg.w + bb.w;
        *(v4f*)(orow + col) = o;
    }
}

__global__ void zero_kernel(float* __restrict__ p, long n) {
    long i = (long)blockIdx.x * blockDim.x + threadIdx.x;
    if (i < n) p[i] = 0.0f;
}

// out[n*K + k] = (f16) in[k*N + n]   (transpose fan_in x fan_out -> rows = out-channel)
__global__ void wcvt_kernel(const float* __restrict__ in, _Float16* __restrict__ out,
                            int K, int N) {
    int i = blockIdx.x * blockDim.x + threadIdx.x;
    if (i >= K * N) return;
    int n = i / K;
    int k = i % K;
    out[n * K + k] = (_Float16)in[k * N + n];
}

extern "C" void kernel_launch(void* const* d_in, const int* in_sizes, int n_in,
                              void* d_out, int out_size, void* d_ws, size_t ws_size,
                              hipStream_t stream) {
    (void)in_sizes; (void)n_in; (void)out_size; (void)ws_size;
    const float* ndata = (const float*)d_in[0];
    const float* edata = (const float*)d_in[1];
    const int*   src   = (const int*)d_in[2];
    const int*   dst   = (const int*)d_in[3];
    const float* eW1 = (const float*)d_in[4],  *eb1   = (const float*)d_in[5];
    const float* eW2 = (const float*)d_in[6],  *eb2   = (const float*)d_in[7];
    const float* egm = (const float*)d_in[8],  *ebeta = (const float*)d_in[9];
    const float* nW1 = (const float*)d_in[10], *nb1   = (const float*)d_in[11];
    const float* nW2 = (const float*)d_in[12], *nb2   = (const float*)d_in[13];
    const float* ngm = (const float*)d_in[14], *nbeta = (const float*)d_in[15];

    char* ws = (char*)d_ws;
    _Float16* eW1T = (_Float16*)(ws + 0);       // 128x192 f16
    _Float16* eW2T = (_Float16*)(ws + 49152);   // 64x128 f16
    _Float16* nW1T = (_Float16*)(ws + 65536);   // 128x192 f16
    _Float16* nW2T = (_Float16*)(ws + 114688);  // 64x128 f16
    float* eu = (float*)(ws + 131072);          // N x 64 sums (src)
    float* ev = eu + (long)N_NODES * 64;        // N x 64 sums (dst)
    float* cu = ev + (long)N_NODES * 64;        // N counts
    float* cv = cu + N_NODES;                   // N counts

    float* out_n = (float*)d_out;               // [N,64]
    float* out_e = out_n + (long)N_NODES * 64;  // [E,64]

    long nz = 2L * N_NODES * 64 + 2L * N_NODES;
    zero_kernel<<<(int)((nz + 255) / 256), 256, 0, stream>>>(eu, nz);

    wcvt_kernel<<<(192 * 128 + 255) / 256, 256, 0, stream>>>(eW1, eW1T, 192, 128);
    wcvt_kernel<<<(128 * 64 + 255) / 256, 256, 0, stream>>>(eW2, eW2T, 128, 64);
    wcvt_kernel<<<(192 * 128 + 255) / 256, 256, 0, stream>>>(nW1, nW1T, 192, 128);
    wcvt_kernel<<<(128 * 64 + 255) / 256, 256, 0, stream>>>(nW2, nW2T, 128, 64);

    edge_kernel<<<(N_EDGES / 16) / 4, 128, 0, stream>>>(
        ndata, edata, src, dst, eb1, eb2, egm, ebeta, eW1T, eW2T,
        out_e, eu, ev, cu, cv);

    node_kernel<<<(N_NODES / 16 + 3) / 4, 128, 0, stream>>>(
        ndata, eu, ev, cu, cv, nb1, nb2, ngm, nbeta, nW1T, nW2T, out_n);
}
